// CRF_78168404787757
// MI455X (gfx1250) — matrix operand order repaired and checked by reading, not compiled, not run on hardware
//
#include <hip/hip_runtime.h>
#include <hip/hip_bf16.h>
#include <stdint.h>

typedef __attribute__((ext_vector_type(16))) _Float16     v16h;
typedef __attribute__((ext_vector_type(8)))  float        v8f;
typedef __attribute__((ext_vector_type(4)))  float        f4v;
typedef __attribute__((ext_vector_type(4)))  unsigned int u32x4;
typedef __attribute__((ext_vector_type(8)))  int          i32x8;
typedef __attribute__((ext_vector_type(4)))  int          i32x4;

#define NTAG  512
#define BATCH 64
#define TLEN  512
#define VSTR  520   // halves: padded row stride of vbuf (1040B = 65*16B, spreads banks)
#define ESTR  512   // floats: dense row stride of ebuf (TDM writes a dense tile)

union ABu { v16h v; f4v q[2]; };

// ---------------- prep: column max of trans ----------------
__global__ void k_colmax(const float* __restrict__ trans, float* __restrict__ cmax) {
  int j = threadIdx.x;
  float m = -1e30f;
  for (int i = 0; i < NTAG; ++i) m = fmaxf(m, trans[i * NTAG + j]);
  cmax[j] = m;
}

// ---- prep: P[i][j] = exp(trans[i][j]-cmax[j]) as f16, in WMMA B-fragment order ----
// fragment f = jt*16 + kt (jt: 16-col tile, kt: 32-K tile); within: lane-major,
// each lane holds 16 consecutive K halves for its column (lanes>=16: K+16).
__global__ void k_packP(const float* __restrict__ trans, const float* __restrict__ cmax,
                        _Float16* __restrict__ P) {
  int id = blockIdx.x * blockDim.x + threadIdx.x;   // 0 .. 512*512-1 (halves)
  int h  = id & 15;
  int l  = (id >> 4) & 31;
  int kt = (id >> 9) & 15;
  int jt = id >> 13;
  int J  = jt * 16 + (l & 15);
  int K  = kt * 32 + ((l >> 4) << 4) + h;
  float v = __expf(trans[K * NTAG + J] - cmax[J]);
  P[id] = (_Float16)v;
}

// Issue a TDM load of a 16 x 512 f32 tile (row stride TLEN*NTAG elements) into LDS.
__device__ __forceinline__ void tdm_load_emits_tile(const float* gsrc, unsigned lds_off) {
  unsigned long long ga = (unsigned long long)(const void*)gsrc;
  u32x4 g0;
  g0.x = 1u;                                            // count=1, user mode
  g0.y = lds_off;                                       // lds_addr (bytes)
  g0.z = (unsigned)(ga & 0xffffffffu);                  // global_addr[31:0]
  g0.w = (unsigned)((ga >> 32) & 0x01ffffffu)           // global_addr[56:32]
         | 0x80000000u;                                 // type=2 ("image")
  i32x8 g1;
  g1[0] = 0x00020000;                                   // data_size=2 (4 bytes)
  g1[1] = (NTAG & 0xffff) << 16;                        // tensor_dim0[15:0]=512
  g1[2] = 16 << 16;                                     // tensor_dim1[15:0]=16
  g1[3] = NTAG << 16;                                   // tile_dim0=512
  g1[4] = 16;                                           // tile_dim1=16, tile_dim2=0
  g1[5] = TLEN * NTAG;                                  // tensor_dim0_stride=262144
  g1[6] = 0;
  g1[7] = 0;
  i32x4 gz = {0, 0, 0, 0};
#if defined(__has_include) && __has_include(<hip/amd_detail/amd_gfx1250_TDM.h>)
  i32x8 gz8 = {0, 0, 0, 0, 0, 0, 0, 0};                 // therock headers: 6-arg builtin
  __builtin_amdgcn_tensor_load_to_lds(g0, g1, gz, gz, gz8, 0);
#else
  __builtin_amdgcn_tensor_load_to_lds(g0, g1, gz, gz, 0);  // ROCm 7.2: 5-arg builtin
#endif
}

// ---------------- forward recursion: exp-domain GEMM via WMMA + TDM staging ------
__global__ __launch_bounds__(1024)
void k_forward(const float* __restrict__ emits, const _Float16* __restrict__ P,
               const float* __restrict__ cmax, float* __restrict__ out) {
  __shared__ _Float16 vbuf[16 * VSTR];   // V = exp(fwd - M), f16, A-matrix source
  __shared__ float    ebuf[16 * ESTR];   // emits tile, written by TDM
  __shared__ float    wred[32][16];
  __shared__ float    Mprev[16];
  __shared__ float    Mrow[16];

  const int tid  = threadIdx.x;
  const int lane = tid & 31;
  const int w    = tid >> 5;             // wave = 16-column tile
  const int hi   = (lane >> 4) & 1;
  const int arow = lane & 15;
  const int jloc = (w << 4) + arow;      // 0..511
  const int b0   = blockIdx.x * 16;

  // LDS byte offset of ebuf (aperture truncation: low 32 bits = LDS offset)
  const unsigned ebuf_lds = (unsigned)(size_t)(void*)ebuf;

  for (int i = tid; i < 16 * VSTR; i += 1024) vbuf[i] = (_Float16)1.0f;  // exp(0-0)
  if (tid < 16) Mprev[tid] = 0.0f;
  __syncthreads();

  const float cj = cmax[jloc];
  float vex[8] = {0.f, 0.f, 0.f, 0.f, 0.f, 0.f, 0.f, 0.f};

  for (int t = 0; t < TLEN; ++t) {
    // wave 0: DMA emits[b0..b0+15][t][0..511] -> LDS via Tensor Data Mover.
    // Overlaps with the whole WMMA loop below; no VGPR traffic.
    if (tid < 32)
      tdm_load_emits_tile(emits + (((size_t)b0) * TLEN + t) * NTAG, ebuf_lds);

    // U(16x16) = V(16x512) x P(512 x 16-col tile), f16 in / f32 acc
    v8f acc = {0.f, 0.f, 0.f, 0.f, 0.f, 0.f, 0.f, 0.f};
    #pragma unroll 4
    for (int kt = 0; kt < 16; ++kt) {
      const int kb = kt << 5;
      ABu a, b;
      // A 16x32 f16 layout: lanes<16: K=kb+0..15; lanes>=16: K=kb+16..31
      a.q[0] = *(const f4v*)(vbuf + arow * VSTR + kb + hi * 16);
      a.q[1] = *(const f4v*)(vbuf + arow * VSTR + kb + hi * 16 + 8);
      const f4v* bp = (const f4v*)P + ((size_t)((w << 4) + kt) * 32 + lane) * 2;
      b.q[0] = bp[0];
      b.q[1] = bp[1];
      acc = __builtin_amdgcn_wmma_f32_16x16x32_f16(false, a.v, false, b.v,
                                                   (short)0, acc, false, false);
    }

    if (tid < 32) __builtin_amdgcn_s_wait_tensorcnt(0);  // tile landed in LDS
    __syncthreads();                                     // publish ebuf; vbuf reads done

    // fwd' = log(u) + Mprev + c_j + e    (D layout: M = r + 8*hi, N = lane&15)
    float fwdv[8];
    #pragma unroll
    for (int r = 0; r < 8; ++r) {
      int row = r + hi * 8;
      fwdv[r] = __logf(acc[r]) + Mprev[row] + cj + ebuf[row * ESTR + jloc];
    }
    // per-row max across this wave's 16 columns (butterfly within 16-lane half)
    #pragma unroll
    for (int r = 0; r < 8; ++r) {
      float m = fwdv[r];
      m = fmaxf(m, __shfl_xor(m, 1, 32));
      m = fmaxf(m, __shfl_xor(m, 2, 32));
      m = fmaxf(m, __shfl_xor(m, 4, 32));
      m = fmaxf(m, __shfl_xor(m, 8, 32));
      if (arow == 0) wred[w][r + hi * 8] = m;
    }
    __syncthreads();
    if (tid < 16) {
      float m = -1e30f;
      #pragma unroll 8
      for (int ww = 0; ww < 32; ++ww) m = fmaxf(m, wred[ww][tid]);
      Mrow[tid]  = m;
      Mprev[tid] = m;
    }
    __syncthreads();
    #pragma unroll
    for (int r = 0; r < 8; ++r) {
      int row = r + hi * 8;
      vex[r] = __expf(fwdv[r] - Mrow[row]);
      vbuf[row * VSTR + jloc] = (_Float16)vex[r];
    }
    __syncthreads();
  }

  // log_Z = M_final + log(sum_j exp(fwd - M_final))
  #pragma unroll
  for (int r = 0; r < 8; ++r) {
    float s = vex[r];
    s += __shfl_xor(s, 1, 32);
    s += __shfl_xor(s, 2, 32);
    s += __shfl_xor(s, 4, 32);
    s += __shfl_xor(s, 8, 32);
    if (arow == 0) wred[w][r + hi * 8] = s;
  }
  __syncthreads();
  if (tid < 16) {
    float tot = 0.f;
    for (int ww = 0; ww < 32; ++ww) tot += wred[ww][tid];
    out[64 + b0 + tid] = Mprev[tid] + __logf(tot);
  }
}

// ---------------- Viterbi (max-plus): VALU, i-loop split across thread pairs ----
__global__ __launch_bounds__(1024)
void k_viterbi(const float* __restrict__ emits, const float* __restrict__ trans,
               unsigned short* __restrict__ bt, int* __restrict__ last,
               float* __restrict__ out) {
  __shared__ float dp[512];
  __shared__ float sv[1024];
  __shared__ int   si[1024];
  const int tid  = threadIdx.x;
  const int b    = blockIdx.x;
  const int j    = tid & 511;
  const int half = tid >> 9;
  const int i0   = half * 256;

  if (tid < 512) dp[tid] = emits[((size_t)b * TLEN) * NTAG + tid];
  __syncthreads();

  for (int t = 1; t < TLEN; ++t) {
    float best = -1e30f;
    int   bi   = i0;
    for (int ii = i0; ii < i0 + 256; ii += 64) {
      if (ii + 64 < NTAG) __builtin_prefetch(trans + (ii + 64) * NTAG + j, 0, 0);
      #pragma unroll 8
      for (int i = ii; i < ii + 64; ++i) {
        float s = dp[i] + trans[i * NTAG + j];
        if (s > best) { best = s; bi = i; }
      }
    }
    sv[tid] = best; si[tid] = bi;
    float e = (tid < 512) ? emits[((size_t)b * TLEN + t) * NTAG + j] : 0.f;
    __syncthreads();
    if (tid < 512) {
      float v0 = sv[tid], v1 = sv[tid + 512];
      int bidx = (v1 > v0) ? si[tid + 512] : si[tid];   // ties -> lower i (lower half)
      float bv = fmaxf(v0, v1);
      bt[((size_t)(t - 1) * BATCH + b) * NTAG + j] = (unsigned short)bidx;
      dp[j] = bv + e;
    }
    __syncthreads();
  }

  if (tid < 512) { sv[tid] = dp[tid]; si[tid] = tid; }
  __syncthreads();
  for (int s = 256; s > 0; s >>= 1) {
    if (tid < s) {
      float a = sv[tid], c = sv[tid + s];
      if (c > a || (c == a && si[tid + s] < si[tid])) { sv[tid] = c; si[tid] = si[tid + s]; }
    }
    __syncthreads();
  }
  if (tid == 0) { out[128 + b] = sv[0]; last[b] = si[0]; }
}

// ---------------- backtrack (serial chain per batch row) ----------------
__global__ void k_backtrack(const unsigned short* __restrict__ bt,
                            const int* __restrict__ last, float* __restrict__ out) {
  int b = threadIdx.x;
  int cur = last[b];
  out[192 + b * TLEN + (TLEN - 1)] = (float)cur;
  for (int k = TLEN - 2; k >= 0; --k) {
    cur = bt[((size_t)k * BATCH + b) * NTAG + cur];
    out[192 + b * TLEN + k] = (float)cur;
  }
}

// ---------------- gold path scores ----------------
__global__ void k_pathscore(const float* __restrict__ emits, const int* __restrict__ tags,
                            const float* __restrict__ trans, float* __restrict__ out) {
  __shared__ float red[512];
  int b = blockIdx.x, t = threadIdx.x;
  int tag = tags[b * TLEN + t];
  float acc = emits[((size_t)b * TLEN + t) * NTAG + tag];
  if (t < TLEN - 1) acc += trans[tag * NTAG + tags[b * TLEN + t + 1]];
  red[t] = acc;
  __syncthreads();
  for (int s = 256; s > 0; s >>= 1) {
    if (t < s) red[t] += red[t + s];
    __syncthreads();
  }
  if (t == 0) out[b] = red[0];
}

extern "C" void kernel_launch(void* const* d_in, const int* in_sizes, int n_in,
                              void* d_out, int out_size, void* d_ws, size_t ws_size,
                              hipStream_t stream) {
  const float* emits = (const float*)d_in[0];
  const int*   tags  = (const int*)d_in[1];
  const float* trans = (const float*)d_in[2];
  float* out = (float*)d_out;   // [0,64): path_scores [64,128): log_Z
                                // [128,192): vit_scores [192,192+64*512): paths

  char* ws = (char*)d_ws;
  _Float16*       P    = (_Float16*)(ws);                // 524288 B
  float*          cmax = (float*)(ws + 524288);          // 2048 B
  int*            last = (int*)(ws + 526336);            // 256 B
  unsigned short* bt   = (unsigned short*)(ws + 526592); // 64*511*512*2 B

  k_pathscore<<<dim3(BATCH), dim3(TLEN), 0, stream>>>(emits, tags, trans, out);
  k_colmax  <<<dim3(1),     dim3(NTAG), 0, stream>>>(trans, cmax);
  k_packP   <<<dim3(512),   dim3(512),  0, stream>>>(trans, cmax, P);
  k_forward <<<dim3(4),     dim3(1024), 0, stream>>>(emits, P, cmax, out);
  k_viterbi <<<dim3(BATCH), dim3(1024), 0, stream>>>(emits, trans, bt, last, out);
  k_backtrack<<<dim3(1),    dim3(BATCH), 0, stream>>>(bt, last, out);
}
